// RelativeAttention_67997922230301
// MI455X (gfx1250) — matrix operand order repaired and checked by reading, not compiled, not run on hardware
//
#include <hip/hip_runtime.h>
#include <hip/hip_bf16.h>

// ---------------------------------------------------------------------------
// Relative-position attention, fused flash-style, f16 WMMA on gfx1250 (wave32)
// B=2, S=2048, D=768, H=12, U=64, M=64
// ---------------------------------------------------------------------------

#define BB 2
#define SS 2048
#define DD 768
#define HH 12
#define UU 64
#define MM 64
#define NREL 129           // 2*M+1
#define NRELP 144          // padded to 9 j-tiles of 16
#define NRELW 160          // padded to 5 K=32 chunks for w @ rel_v

typedef __attribute__((ext_vector_type(16))) _Float16 v16h;
typedef __attribute__((ext_vector_type(8)))  _Float16 v8h;
typedef __attribute__((ext_vector_type(8)))  float    v8f;
typedef __attribute__((ext_vector_type(4)))  float    v4f;

__device__ __forceinline__ v8f wmma_f16(v16h a, v16h b, v8f c) {
  // D = A(16x32 f16) * B(32x16 f16) + C(16x16 f32)
  return __builtin_amdgcn_wmma_f32_16x16x32_f16(
      /*neg_a=*/false, a, /*neg_b=*/false, b,
      /*c_mod=*/(short)0, c, /*reuse_a=*/false, /*reuse_b=*/false);
}

// Build a v16h A/B fragment from two contiguous 16B half-chunks.
__device__ __forceinline__ v16h frag2(const _Float16* p0, const _Float16* p1) {
  union { v16h v; v8h h[2]; } u;
  u.h[0] = *(const v8h*)p0;
  u.h[1] = *(const v8h*)p1;
  return u.v;
}

// Build a v16h A fragment from two 8-float chunks (fp32 -> f16 convert).
__device__ __forceinline__ v16h frag2f(const float* p0, const float* p1) {
  union { v16h v; _Float16 e[16]; } u;
#pragma unroll
  for (int i = 0; i < 8; ++i) {
    u.e[i]     = (_Float16)p0[i];
    u.e[8 + i] = (_Float16)p1[i];
  }
  return u.v;
}

__device__ __forceinline__ void lds_fence() {
  asm volatile("s_wait_dscnt 0" ::: "memory");
}

// 16-lane reductions: masks < 16 keep data inside each half-wave, which is
// exactly where one C-matrix row lives (lanes 0-15 hold rows 0-7, 16-31 rows 8-15).
__device__ __forceinline__ float red_max16(float x) {
#pragma unroll
  for (int m = 1; m < 16; m <<= 1) x = fmaxf(x, __shfl_xor(x, m, 32));
  return x;
}
__device__ __forceinline__ float red_sum16(float x) {
#pragma unroll
  for (int m = 1; m < 16; m <<= 1) x += __shfl_xor(x, m, 32);
  return x;
}

// ---------------------------------------------------------------------------
// Kernel 0: weight transpose + f16 convert, rel-embedding pad/transpose.
//   Wt[mat][n][k]   = (f16) W[mat][k][n]           (3 x 768 x 768)
//   relkp[j][u]     = (f16) rel_k[j][u], j<129 else 0   (144 x 64)
//   RvT[u][j]       = (f16) rel_v[j][u], j<129 else 0   (64 x 160)
// ---------------------------------------------------------------------------
__global__ void prep_kernel(const float* __restrict__ Wq, const float* __restrict__ Wk,
                            const float* __restrict__ Wv, const float* __restrict__ relk,
                            const float* __restrict__ relv,
                            _Float16* __restrict__ Wt, _Float16* __restrict__ relkp,
                            _Float16* __restrict__ RvT) {
  const int WE = DD * DD;
  int idx = blockIdx.x * blockDim.x + threadIdx.x;
  if (idx < 3 * WE) {
    int mat = idx / WE;
    int r   = idx % WE;
    int n = r / DD, k = r % DD;
    const float* W = (mat == 0) ? Wq : (mat == 1) ? Wk : Wv;
    Wt[(size_t)mat * WE + (size_t)n * DD + k] = (_Float16)W[(size_t)k * DD + n];
  } else if (idx < 3 * WE + NRELP * UU) {
    int r = idx - 3 * WE;
    int j = r / UU, u = r % UU;
    relkp[j * UU + u] = (j < NREL) ? (_Float16)relk[j * UU + u] : (_Float16)0.f;
  } else if (idx < 3 * WE + NRELP * UU + UU * NRELW) {
    int r = idx - 3 * WE - NRELP * UU;
    int u = r / NRELW, j = r % NRELW;
    RvT[u * NRELW + j] = (j < NREL) ? (_Float16)relv[j * UU + u] : (_Float16)0.f;
  }
}

// ---------------------------------------------------------------------------
// Kernel 1: QKV projection.  One wave per 16x16 output tile, K = 768 = 24 WMMAs.
//   X:  [B*S, 768] fp32 (A operand, converted to f16 on load)
//   Wt: [768 n, 768 k] f16 (B operand, 2 x b128 per lane per K-chunk)
//   Q/K out: [B,H,S,U] f16 row-major;  V out: [B,H,U,S] f16 (transposed)
// ---------------------------------------------------------------------------
__global__ __launch_bounds__(32) void proj_kernel(
    const float* __restrict__ X, const _Float16* __restrict__ Wt,
    const float* __restrict__ bq, const float* __restrict__ bk,
    const float* __restrict__ bv,
    _Float16* __restrict__ Qh, _Float16* __restrict__ Kh,
    _Float16* __restrict__ VhT) {
  const int lane = threadIdx.x & 31;
  const int hi = lane >> 4, ln = lane & 15;
  const int mt = blockIdx.x, nt = blockIdx.y, mat = blockIdx.z;

  const _Float16* W   = Wt + (size_t)mat * DD * DD;
  const float* bias   = (mat == 0) ? bq : (mat == 1) ? bk : bv;
  const float* xrow   = X + (size_t)(mt * 16 + ln) * DD;   // A row m = ln (both halves)
  const _Float16* wrow = W + (size_t)(nt * 16 + ln) * DD;  // B col n = ln

  v8f c = {};
#pragma unroll 4
  for (int kk = 0; kk < DD; kk += 32) {
    // A f16 layout: lane group hi picks K subset {0..15} / {16..31}
    v16h a = frag2f(xrow + kk + 16 * hi, xrow + kk + 16 * hi + 8);
    // B f16 layout: lane group hi picks K subset {0..15} / {16..31}
    v16h b = frag2(wrow + kk + hi * 16, wrow + kk + hi * 16 + 8);
    c = wmma_f16(a, b, c);
  }

  const int n = nt * 16 + ln;
  const float bsc = bias[n];
  const int h = n >> 6, u = n & 63;
#pragma unroll
  for (int v = 0; v < 8; ++v) {
    float val = c[v] + bsc;
    int tok = mt * 16 + v + 8 * hi;          // C layout: row = v + 8*hi
    int b0 = tok >> 11, s0 = tok & (SS - 1);
    size_t bh = (size_t)(b0 * HH + h);
    if (mat == 0)      Qh[(bh * SS + s0) * UU + u] = (_Float16)val;
    else if (mat == 1) Kh[(bh * SS + s0) * UU + u] = (_Float16)val;
    else               VhT[(bh * UU + u) * SS + s0] = (_Float16)val;
  }
}

// ---------------------------------------------------------------------------
// Kernel 2: fused relative attention.  One wave per (b, h, 16-query tile).
// Online softmax over 32-key chunks; rel-k via per-tile R table (WMMA),
// rel-v via per-row probability histogram w[129] (LDS atomics) + final WMMA.
// ---------------------------------------------------------------------------
__global__ __launch_bounds__(32) void attn_kernel(
    const _Float16* __restrict__ Qh, const _Float16* __restrict__ Kh,
    const _Float16* __restrict__ VhT, const _Float16* __restrict__ relkp,
    const _Float16* __restrict__ RvT, const int* __restrict__ mask,
    float* __restrict__ out) {
  const int lane = threadIdx.x & 31;
  const int hi = lane >> 4, ln = lane & 15;
  const int bh = blockIdx.y;
  const int b = bh / HH, h = bh % HH;
  const int qbase = blockIdx.x * 16;

  __shared__ float    Rtab[16 * NRELP];   // R[m][j] = q_m . rel_k_emb[j]
  __shared__ float    wacc[16 * NRELW];   // w[m][j] = sum of probs with idx j
  __shared__ _Float16 pst[16 * 32];       // probs staging: C-layout -> A-layout

  for (int i = lane; i < 16 * NRELW; i += 32) wacc[i] = 0.f;

  // Q fragments (A operand): row m = ln, K chunks u 0..31 / 32..63
  const _Float16* qrow = Qh + ((size_t)bh * SS + qbase + ln) * UU;
  const v16h qa0 = frag2(qrow + 16 * hi, qrow + 8 + 16 * hi);
  const v16h qa1 = frag2(qrow + 32 + 16 * hi, qrow + 40 + 16 * hi);

  // --- R table: [16 x 144] = Q_tile @ rel_k_emb^T, 9 j-tiles x 2 WMMA ---
#pragma unroll
  for (int jt = 0; jt < 9; ++jt) {
    const _Float16* rr = relkp + (size_t)(jt * 16 + ln) * UU + hi * 16;
    v16h rb0 = frag2(rr, rr + 8);
    v16h rb1 = frag2(rr + 32, rr + 40);
    v8f c = {};
    c = wmma_f16(qa0, rb0, c);
    c = wmma_f16(qa1, rb1, c);
#pragma unroll
    for (int v = 0; v < 8; ++v) Rtab[(v + 8 * hi) * NRELP + jt * 16 + ln] = c[v];
  }
  lds_fence();

  int mq[8];
#pragma unroll
  for (int v = 0; v < 8; ++v) mq[v] = mask[b * SS + qbase + v + 8 * hi];

  float mrow[8], lrow[8];
#pragma unroll
  for (int v = 0; v < 8; ++v) { mrow[v] = -1e30f; lrow[v] = 0.f; }
  v8f acc0 = {}, acc1 = {}, acc2 = {}, acc3 = {};

  for (int kb = 0; kb < SS; kb += 32) {
    float s[2][8], p[2][8];

    // --- scores for two 16-key subtiles ---
#pragma unroll
    for (int t = 0; t < 2; ++t) {
      const int key = kb + t * 16 + ln;  // this lane's key column (C layout N)
      const _Float16* krow = Kh + ((size_t)bh * SS + key) * UU + hi * 16;
      v16h kb0 = frag2(krow, krow + 8);          // u 0..31
      v16h kb1 = frag2(krow + 32, krow + 40);    // u 32..63
      v8f c = {};
      c = wmma_f16(qa0, kb0, c);
      c = wmma_f16(qa1, kb1, c);
      const int mk = mask[b * SS + key];
#pragma unroll
      for (int v = 0; v < 8; ++v) {
        const int m = v + 8 * hi;
        int d = key - (qbase + m);
        d = (d < -MM) ? -MM : (d > MM) ? MM : d;
        float sc = (c[v] + Rtab[m * NRELP + d + MM]) * 0.125f;  // 1/sqrt(64)
        sc += (1.0f - (float)(mq[v] * mk)) * -10000.0f;
        s[t][v] = sc;
      }
    }

    // --- online softmax statistics (per-row, half-wave reductions) ---
    float mnew[8], scl[8], rs[8];
#pragma unroll
    for (int v = 0; v < 8; ++v) {
      float cm = red_max16(fmaxf(s[0][v], s[1][v]));
      mnew[v] = fmaxf(mrow[v], cm);
      scl[v] = __expf(mrow[v] - mnew[v]);
    }
#pragma unroll
    for (int t = 0; t < 2; ++t)
#pragma unroll
      for (int v = 0; v < 8; ++v) {
        float pp = __expf(s[t][v] - mnew[v]);
        p[t][v] = pp;
        pst[(v + 8 * hi) * 32 + t * 16 + ln] = (_Float16)pp;
      }
#pragma unroll
    for (int v = 0; v < 8; ++v) {
      rs[v] = red_sum16(p[0][v] + p[1][v]);
      lrow[v] = lrow[v] * scl[v] + rs[v];
      acc0[v] *= scl[v]; acc1[v] *= scl[v]; acc2[v] *= scl[v]; acc3[v] *= scl[v];
      mrow[v] = mnew[v];
    }

    // rescale w rows whose max moved (each (row,col) touched by exactly one lane)
#pragma unroll
    for (int v = 0; v < 8; ++v) {
      if (scl[v] < 1.0f) {
        float* wr = &wacc[(v + 8 * hi) * NRELW];
        for (int j = ln; j < NRELW; j += 16) wr[j] *= scl[v];
      }
    }

    // --- rel-v histogram accumulation ---
    if (kb + 31 <= qbase - MM) {
      // whole chunk clips to idx 0 for every row: fold row sums
      if (ln == 0) {
#pragma unroll
        for (int v = 0; v < 8; ++v) atomicAdd(&wacc[(v + 8 * hi) * NRELW + 0], rs[v]);
      }
    } else if (kb >= qbase + 15 + MM) {
      // whole chunk clips to idx 2M
      if (ln == 0) {
#pragma unroll
        for (int v = 0; v < 8; ++v) atomicAdd(&wacc[(v + 8 * hi) * NRELW + 2 * MM], rs[v]);
      }
    } else {
#pragma unroll
      for (int t = 0; t < 2; ++t)
#pragma unroll
        for (int v = 0; v < 8; ++v) {
          const int m = v + 8 * hi;
          int d = (kb + t * 16 + ln) - (qbase + m);
          d = (d < -MM) ? -MM : (d > MM) ? MM : d;
          atomicAdd(&wacc[m * NRELW + d + MM], p[t][v]);
        }
    }
    lds_fence();

    // --- P @ V: probs A-fragment from LDS, V^T B-fragments from global ---
    v16h pa = frag2(&pst[ln * 32 + 16 * hi], &pst[ln * 32 + 8 + 16 * hi]);
    {
      const _Float16* v0 = VhT + ((size_t)bh * UU + 0 * 16 + ln) * SS + kb + hi * 16;
      const _Float16* v1 = VhT + ((size_t)bh * UU + 1 * 16 + ln) * SS + kb + hi * 16;
      const _Float16* v2 = VhT + ((size_t)bh * UU + 2 * 16 + ln) * SS + kb + hi * 16;
      const _Float16* v3 = VhT + ((size_t)bh * UU + 3 * 16 + ln) * SS + kb + hi * 16;
      acc0 = wmma_f16(pa, frag2(v0, v0 + 8), acc0);
      acc1 = wmma_f16(pa, frag2(v1, v1 + 8), acc1);
      acc2 = wmma_f16(pa, frag2(v2, v2 + 8), acc2);
      acc3 = wmma_f16(pa, frag2(v3, v3 + 8), acc3);
    }
    lds_fence();
  }

  // --- rel-v term: acc += w[16 x 160] @ RvT (5 K-chunks x 4 u-tiles) ---
  lds_fence();
#pragma unroll
  for (int kc = 0; kc < 5; ++kc) {
    const int jb = kc * 32;
    const float* wr = &wacc[ln * NRELW + jb];
    v16h wa = frag2f(wr + 16 * hi, wr + 8 + 16 * hi);
    const _Float16* r0 = RvT + (size_t)(0 * 16 + ln) * NRELW + jb + hi * 16;
    const _Float16* r1 = RvT + (size_t)(1 * 16 + ln) * NRELW + jb + hi * 16;
    const _Float16* r2 = RvT + (size_t)(2 * 16 + ln) * NRELW + jb + hi * 16;
    const _Float16* r3 = RvT + (size_t)(3 * 16 + ln) * NRELW + jb + hi * 16;
    acc0 = wmma_f16(wa, frag2(r0, r0 + 8), acc0);
    acc1 = wmma_f16(wa, frag2(r1, r1 + 8), acc1);
    acc2 = wmma_f16(wa, frag2(r2, r2 + 8), acc2);
    acc3 = wmma_f16(wa, frag2(r3, r3 + 8), acc3);
  }

  // --- normalize and store: out[b, s, h*64 + u] fp32 ---
#pragma unroll
  for (int v = 0; v < 8; ++v) {
    const float inv = 1.0f / lrow[v];
    const int row = qbase + v + 8 * hi;
    float* orow = out + ((size_t)b * SS + row) * (HH * UU) + h * UU;
    orow[0 * 16 + ln] = acc0[v] * inv;
    orow[1 * 16 + ln] = acc1[v] * inv;
    orow[2 * 16 + ln] = acc2[v] * inv;
    orow[3 * 16 + ln] = acc3[v] * inv;
  }
}

// ---------------------------------------------------------------------------
extern "C" void kernel_launch(void* const* d_in, const int* in_sizes, int n_in,
                              void* d_out, int out_size, void* d_ws, size_t ws_size,
                              hipStream_t stream) {
  const float* X    = (const float*)d_in[0];
  const int*   mask = (const int*)d_in[1];
  const float* Wq   = (const float*)d_in[2];
  const float* bq   = (const float*)d_in[3];
  const float* Wk   = (const float*)d_in[4];
  const float* bk   = (const float*)d_in[5];
  const float* Wv   = (const float*)d_in[6];
  const float* bv   = (const float*)d_in[7];
  const float* relk = (const float*)d_in[8];
  const float* relv = (const float*)d_in[9];
  float* out = (float*)d_out;

  // workspace layout (f16)
  char* w = (char*)d_ws;
  _Float16* Wt    = (_Float16*)w;  w += (size_t)3 * DD * DD * 2;
  _Float16* relkp = (_Float16*)w;  w += (size_t)NRELP * UU * 2;
  _Float16* RvT   = (_Float16*)w;  w += (size_t)UU * NRELW * 2;
  _Float16* Qh    = (_Float16*)w;  w += (size_t)BB * HH * SS * UU * 2;
  _Float16* Kh    = (_Float16*)w;  w += (size_t)BB * HH * SS * UU * 2;
  _Float16* VhT   = (_Float16*)w;  w += (size_t)BB * HH * SS * UU * 2;

  // 0) weight transpose/convert + rel-embedding prep
  const int prep_total = 3 * DD * DD + NRELP * UU + UU * NRELW;
  prep_kernel<<<(prep_total + 255) / 256, 256, 0, stream>>>(
      Wq, Wk, Wv, relk, relv, Wt, relkp, RvT);

  // 1) QKV projection: 16x16 tiles over [B*S=4096, H*U=768], 3 matrices
  proj_kernel<<<dim3((BB * SS) / 16, DD / 16, 3), 32, 0, stream>>>(
      X, Wt, bq, bk, bv, Qh, Kh, VhT);

  // 2) fused relative attention: one wave per (q-tile, b*h)
  attn_kernel<<<dim3(SS / 16, BB * HH), 32, 0, stream>>>(
      Qh, Kh, VhT, relkp, RvT, mask, out);
}
